// CRF_69114613730357
// MI455X (gfx1250) — matrix operand order repaired and checked by reading, not compiled, not run on hardware
//
#include <hip/hip_runtime.h>
#include <hip/hip_bf16.h>

// ---------------------------------------------------------------------------
// CRF loss on MI455X (gfx1250, wave32, WMMA + Tensor Data Mover).
//
// forward recurrence rewritten as matmul:
//   alpha_new = log( E @ exp(alpha - m) ) + m + f_t,   E = exp(transitions)
// E precomputed once into v_wmma f16 A-fragments; per timestep each wave does
// D(64x16) = E(64x64) @ V(64x16) with 8 chained v_wmma_f32_16x16x32_f16.
// feats tiles (16 batches x 64 tags, row stride T*K) are streamed by the TDM
// (tensor_load_to_lds) into double-buffered LDS, paced with s_wait_tensorcnt.
// Last timestep is peeled so the steady-state loop is branch-free.
// ---------------------------------------------------------------------------

typedef __attribute__((ext_vector_type(16))) _Float16 v16h;
typedef __attribute__((ext_vector_type(8)))  float    v8f;
typedef __attribute__((ext_vector_type(4)))  unsigned v4u;
typedef __attribute__((ext_vector_type(8)))  int      v8i;
typedef __attribute__((ext_vector_type(4)))  int      v4i;

#define CRF_B 256
#define CRF_T 1024
#define CRF_K 64
#define CRF_NEG (-10000.0f)

#if defined(__has_builtin)
#if __has_builtin(__builtin_amdgcn_tensor_load_to_lds) && \
    __has_builtin(__builtin_amdgcn_s_wait_tensorcnt)
#define CRF_USE_TDM 1
#endif
#endif
#ifndef CRF_USE_TDM
#define CRF_USE_TDM 0
#endif

__global__ __launch_bounds__(32)
void crf_forward_kernel(const float* __restrict__ feats,
                        const float* __restrict__ trans,
                        const float* __restrict__ masks,
                        float* __restrict__ fwd) {
    const int lane = threadIdx.x;     // 0..31
    const int n    = lane & 15;       // batch-in-tile == N column of C/D
    const int hv   = lane >> 4;       // half-wave select
    const int b0   = blockIdx.x * 16; // batch tile base (grid = B/16 = 16)

    __shared__ float s_alpha[16 * CRF_K];       // [batch-in-tile][tag], 4 KB
#if CRF_USE_TDM
    __shared__ float s_feat[2][16 * CRF_K];     // double-buffered TDM tiles, 8 KB
#endif

    // ---- Precompute E = exp(transitions) in 16x32 f16 A-fragment layout ----
    // A element e of v16h at lane l holds A[M = l%16][K = 16*(e/8) + 8*(l/16) + e%8]
    v16h a_frag[4][2];
#pragma unroll
    for (int it = 0; it < 4; ++it)
#pragma unroll
        for (int kt = 0; kt < 2; ++kt)
#pragma unroll
            for (int e = 0; e < 16; ++e) {
                int i = 16 * it + n;
                int j = 32 * kt + 16 * (e >> 3) + 8 * hv + (e & 7);
                a_frag[it][kt][e] = (_Float16)__expf(trans[i * CRF_K + j]);
            }

    // ---- alpha in C/D layout: lane l, reg r <-> tag = 16*at + 8*hv + r ----
    v8f alpha[4];
#pragma unroll
    for (int at = 0; at < 4; ++at)
#pragma unroll
        for (int r = 0; r < 8; ++r) {
            int tag = 16 * at + 8 * hv + r;
            alpha[at][r] = (tag == 0) ? 0.0f : CRF_NEG;  // START == 0
        }

    const float* frow  = feats + (size_t)(b0 + n) * CRF_T * CRF_K;
    const float* mrowp = masks + (size_t)(b0 + n) * CRF_T;

#if CRF_USE_TDM
    // Constant part of the D#: group1 describes a 2D tile of 4-byte elements,
    // tile 64 x 16 rows out of a tensor with row stride T*K elements.
    v8i g1;
    g1[0] = 0x20000;                 // [17:16] data_size = 2 -> 4 bytes
    g1[1] = (int)(64u << 16);        // tensor_dim0[15:0]  = 64 (bits 79:48)
    g1[2] = (int)(16u << 16);        // tensor_dim1[15:0]  = 16 (bits 111:80)
    g1[3] = (int)(64u << 16);        // tile_dim0          = 64 (bits 127:112)
    g1[4] = 16;                      // tile_dim1 = 16, tile_dim2 = 0
    g1[5] = CRF_T * CRF_K;           // tensor_dim0_stride[31:0] = 65536
    g1[6] = 0;                       // stride[47:32] = 0, dim1_stride lo = 0
    g1[7] = 0;
    const v4i gz  = {0, 0, 0, 0};               // groups 2/3 unused (<=2D)
    const v8i gz8 = {0, 0, 0, 0, 0, 0, 0, 0};   // extra group (clang-23 form)

    auto tdm_issue = [&](int tt, int buf) {
        size_t ga = (size_t)(feats + ((size_t)b0 * CRF_T + tt) * CRF_K);
        unsigned lds_addr = (unsigned)(size_t)(const void*)&s_feat[buf][0];
        v4u g0;
        g0.x = 1u;                                        // count = 1 (valid)
        g0.y = lds_addr;                                  // LDS byte address
        g0.z = (unsigned)ga;                              // global_addr[31:0]
        g0.w = ((unsigned)(ga >> 32) & 0x1FFFFFFu) |      // global_addr[56:32]
               (2u << 30);                                // type = 2 ("image")
        __builtin_amdgcn_tensor_load_to_lds(g0, g1, gz, gz, gz8, 0);
    };

    tdm_issue(1, 0);                 // prologue: fetch tile for t = 1
#endif

    // One recurrence step; `last` is a compile-time constant at both call
    // sites, so the tensorcnt wait selection folds to a single instruction.
    auto step = [&](int t, bool last) __attribute__((always_inline)) {
#if CRF_USE_TDM
        const int buf = (t - 1) & 1;
        if (!last) tdm_issue(t + 1, buf ^ 1);  // prefetch next tile
#else
        if (!last)
            __builtin_prefetch(frow + (size_t)(t + 1) * CRF_K, 0, 0);
#endif

        // per-lane max over this lane's 32 alpha values + dump alpha to LDS
        float mloc = -3.0e38f;
#pragma unroll
        for (int at = 0; at < 4; ++at) {
#pragma unroll
            for (int r = 0; r < 8; ++r) mloc = fmaxf(mloc, alpha[at][r]);
            float* dst = &s_alpha[n * CRF_K + 16 * at + 8 * hv];
            *(float4*)(dst)     = make_float4(alpha[at][0], alpha[at][1], alpha[at][2], alpha[at][3]);
            *(float4*)(dst + 4) = make_float4(alpha[at][4], alpha[at][5], alpha[at][6], alpha[at][7]);
        }
        // combine the two half-waves holding the same batch: max over all 64 tags
        float mrow = fmaxf(mloc, __shfl_xor(mloc, 16, 32));
        __syncthreads();  // single-wave WG: barrier ops degrade to S_NOP

        // ---- build V = exp(alpha - m) in 32x16 f16 B-fragment layout ----
        // B element e of v16h at lane l holds B[K = 16*(e/8) + 8*(l/16) + e%8][N = l%16]
        v16h bfrag[2];
#pragma unroll
        for (int kt = 0; kt < 2; ++kt)
#pragma unroll
            for (int e = 0; e < 16; ++e) {
                float a = s_alpha[n * CRF_K + 32 * kt + 16 * (e >> 3) + 8 * hv + (e & 7)];
                bfrag[kt][e] = (_Float16)__expf(a - mrow);
            }

        // ---- D(64x16) = E @ V : 8x v_wmma_f32_16x16x32_f16, f32 accumulate ----
        v8f d[4];
#pragma unroll
        for (int at = 0; at < 4; ++at) {
            v8f c = {};
            c = __builtin_amdgcn_wmma_f32_16x16x32_f16(
                    false, a_frag[at][0], false, bfrag[0], (short)0, c, false, false);
            c = __builtin_amdgcn_wmma_f32_16x16x32_f16(
                    false, a_frag[at][1], false, bfrag[1], (short)0, c, false, false);
            d[at] = c;
        }

        // ---- wait for this step's feats tile, then update alpha ----
        const float mval = mrowp[t];
#if CRF_USE_TDM
        if (last) __builtin_amdgcn_s_wait_tensorcnt(0);  // final tile done
        else      __builtin_amdgcn_s_wait_tensorcnt(1);  // tile t done, t+1 in flight
        const float* ft = &s_feat[buf][n * CRF_K];
#else
        const float* ft = frow + (size_t)t * CRF_K;
#endif
#pragma unroll
        for (int at = 0; at < 4; ++at) {
            float4 f0 = *(const float4*)&ft[16 * at + 8 * hv];
            float4 f1 = *(const float4*)&ft[16 * at + 8 * hv + 4];
            float fv[8] = {f0.x, f0.y, f0.z, f0.w, f1.x, f1.y, f1.z, f1.w};
#pragma unroll
            for (int r = 0; r < 8; ++r) {
                // clamp avoids log(0) = -inf -> NaN via (1-mask)*(-inf)*0
                float nv = __logf(fmaxf(d[at][r], 1e-30f)) + mrow + fv[r];
                alpha[at][r] = nv * mval + alpha[at][r] * (1.0f - mval);
            }
        }
        __syncthreads();
    };

    for (int t = 1; t < CRF_T - 1; ++t) step(t, false);  // branch-free steady state
    step(CRF_T - 1, true);                               // peeled epilogue

    // ---- forward score: logsumexp over the 64 tags of each batch ----
    float mloc = -3.0e38f;
#pragma unroll
    for (int at = 0; at < 4; ++at)
#pragma unroll
        for (int r = 0; r < 8; ++r) mloc = fmaxf(mloc, alpha[at][r]);
    float mm = fmaxf(mloc, __shfl_xor(mloc, 16, 32));

    float sloc = 0.0f;
#pragma unroll
    for (int at = 0; at < 4; ++at)
#pragma unroll
        for (int r = 0; r < 8; ++r) sloc += __expf(alpha[at][r] - mm);
    float ss = sloc + __shfl_xor(sloc, 16, 32);

    if (hv == 0) fwd[b0 + n] = mm + __logf(ss);
}

// ---- gold path score: embarrassingly parallel, one thread per batch ----
__global__ __launch_bounds__(256)
void crf_gold_kernel(const float* __restrict__ feats,
                     const float* __restrict__ trans,
                     const long long* __restrict__ tags,
                     const float* __restrict__ masks,
                     float* __restrict__ gold) {
    int b = threadIdx.x;
    if (b >= CRF_B) return;
    const long long* tg = tags  + (size_t)b * CRF_T;
    const float*     mk = masks + (size_t)b * CRF_T;
    const float*     fb = feats + (size_t)b * CRF_T * CRF_K;
    float s = 0.0f;
    long long prev = tg[0];
    for (int t = 1; t < CRF_T; ++t) {
        long long cur = tg[t];
        float v = trans[(int)cur * CRF_K + (int)prev] + fb[(size_t)t * CRF_K + (int)cur];
        s += v * mk[t];
        prev = cur;
    }
    gold[b] = s;
}

// ---- mean(forward - gold) ----
__global__ __launch_bounds__(256)
void crf_reduce_kernel(const float* __restrict__ fwd,
                       const float* __restrict__ gold,
                       float* __restrict__ out) {
    __shared__ float sh[CRF_B];
    int i = threadIdx.x;
    sh[i] = fwd[i] - gold[i];
    __syncthreads();
    for (int s = CRF_B / 2; s > 0; s >>= 1) {
        if (i < s) sh[i] += sh[i + s];
        __syncthreads();
    }
    if (i == 0) out[0] = sh[0] * (1.0f / CRF_B);
}

extern "C" void kernel_launch(void* const* d_in, const int* in_sizes, int n_in,
                              void* d_out, int out_size, void* d_ws, size_t ws_size,
                              hipStream_t stream) {
    const float*     feats = (const float*)d_in[0];
    const float*     trans = (const float*)d_in[1];
    const long long* tags  = (const long long*)d_in[2];
    const float*     masks = (const float*)d_in[3];
    float* out  = (float*)d_out;
    float* fwd  = (float*)d_ws;         // [256]
    float* gold = fwd + CRF_B;          // [256]

    crf_forward_kernel<<<dim3(CRF_B / 16), dim3(32), 0, stream>>>(feats, trans, masks, fwd);
    crf_gold_kernel<<<dim3(1), dim3(256), 0, stream>>>(feats, trans, tags, masks, gold);
    crf_reduce_kernel<<<dim3(1), dim3(256), 0, stream>>>(fwd, gold, out);
}
